// AttSplineConv_16149077033176
// MI455X (gfx1250) — matrix operand order repaired and checked by reading, not compiled, not run on hardware
//
#include <hip/hip_runtime.h>

typedef __attribute__((ext_vector_type(16))) _Float16 v16h;
typedef __attribute__((ext_vector_type(8)))  float    v8f;

#define NNODES   100000
#define NEDGES   1600000
#define FDIM     32
#define KSZ      5
#define KTOT     25
#define NEG_SLOPE 0.2f
#define ETILES   (NEDGES / 16)   // 100000, exact
#define NTILES   (NNODES / 16)   // 6250,  exact

#define SCAN_BLOCKS 256
#define CHUNK ((NNODES + SCAN_BLOCKS - 1) / SCAN_BLOCKS)   // 391

// ---------- helpers ----------

__device__ __forceinline__ float red16(float v) {
    v += __shfl_xor(v, 1, 16);
    v += __shfl_xor(v, 2, 16);
    v += __shfl_xor(v, 4, 16);
    v += __shfl_xor(v, 8, 16);
    return v;
}

// Load the 16-f16 A-operand fragment for one row: lane group "hi" (0/1)
// holds K = {hi*8..hi*8+7} U {hi*8+16..hi*8+23} (ISA 16-bit A 16x32 layout).
__device__ __forceinline__ v16h load_a_row(const float* __restrict__ rowp, int hi) {
    const float4* p = (const float4*)(rowp + hi * 8);
    float4 q0 = p[0], q1 = p[1];   // K base .. base+7
    float4 q2 = p[4], q3 = p[5];   // K base+16 .. base+23
    v16h a;
    a[0]  = (_Float16)q0.x;  a[1]  = (_Float16)q0.y;
    a[2]  = (_Float16)q0.z;  a[3]  = (_Float16)q0.w;
    a[4]  = (_Float16)q1.x;  a[5]  = (_Float16)q1.y;
    a[6]  = (_Float16)q1.z;  a[7]  = (_Float16)q1.w;
    a[8]  = (_Float16)q2.x;  a[9]  = (_Float16)q2.y;
    a[10] = (_Float16)q2.z;  a[11] = (_Float16)q2.w;
    a[12] = (_Float16)q3.x;  a[13] = (_Float16)q3.y;
    a[14] = (_Float16)q3.z;  a[15] = (_Float16)q3.w;
    return a;
}

// ---------- kernel 1: x_root = x @ root_weight (WMMA), out = x_root + bias,
//                      rootdot[n] = x_root[n] . att_weight[:32] ----------

__global__ __launch_bounds__(256)
void k_xroot(const float* __restrict__ x, const float* __restrict__ rootw,
             const float* __restrict__ attw, const float* __restrict__ bias,
             float* __restrict__ out, float* __restrict__ rootdot)
{
    // B operand in LDS, f16. Layout: [half(2)][lane(32)][slot(16)]
    // lane<16: col=lane, K=slot<8?slot:8+slot ; lane>=16: col=lane-16, K=slot<8?8+slot:16+slot
    __shared__ __align__(32) _Float16 rl[1024];
    for (int i = threadIdx.x; i < 1024; i += 256) {
        int h  = i >> 9;
        int r2 = i & 511;
        int ln = r2 >> 4;
        int sl = r2 & 15;
        int col = h * 16 + (ln & 15);
        int kk  = ((ln >= 16) ? 8 : 0) + (sl & 7) + ((sl >> 3) << 4);
        rl[i] = (_Float16)rootw[kk * FDIM + col];
    }
    __syncthreads();

    const int lane = threadIdx.x & 31;
    const int wave = threadIdx.x >> 5;
    const int el = lane & 15;
    const int hi = lane >> 4;

    int tile = blockIdx.x * 8 + wave;
    if (tile >= NTILES) return;           // wave-uniform
    const int tb = tile * 16;

    v16h a = load_a_row(x + (size_t)(tb + el) * FDIM, hi);
    v16h b0 = *(const v16h*)(rl + lane * 16);
    v16h b1 = *(const v16h*)(rl + 512 + lane * 16);

    v8f c0 = {}; v8f c1 = {};
    c0 = __builtin_amdgcn_wmma_f32_16x16x32_f16(false, a, false, b0, (short)0, c0, false, false);
    c1 = __builtin_amdgcn_wmma_f32_16x16x32_f16(false, a, false, b1, (short)0, c1, false, false);

    const float aw_lo = attw[el];
    const float aw_hi = attw[16 + el];
    const float b_lo = bias[el];
    const float b_hi = bias[16 + el];

    float myrd = 0.f;
#pragma unroll
    for (int r = 0; r < 8; ++r) {
        int m = r + hi * 8;
        size_t n2 = (size_t)(tb + m) * FDIM;
        out[n2 + el]      = c0[r] + b_lo;
        out[n2 + 16 + el] = c1[r] + b_hi;
        float t = c0[r] * aw_lo + c1[r] * aw_hi;
        t = red16(t);
        if (el == r) myrd = t;
    }
    if (el < 8) rootdot[tb + el + hi * 8] = myrd;
}

// ---------- kernel 2: edge messages via WMMA over 25 spline kernels ----------
// msg_tile (16 edges x 32 out) = sum_k (diag(beta_k) * X[col]) @ W_k
// beta folded into A rows -> C accumulates across all k, no readout per k.
// Also emits the leaky-relu'd attention logit alpha[e].

__global__ __launch_bounds__(256)
void k_edge_msg(const float* __restrict__ x,
                const int* __restrict__ erow, const int* __restrict__ ecol,
                const float* __restrict__ pseudo,
                const float* __restrict__ weight,
                const float* __restrict__ attw,
                const float* __restrict__ rootdot,
                _Float16* __restrict__ msg,
                float* __restrict__ alpha)
{
    // all 25 W matrices in f16, B-operand layout per k: [half(2)][lane(32)][slot(16)]
    __shared__ __align__(32) _Float16 wl[KTOT * 1024];   // 50 KB
    for (int i = threadIdx.x; i < KTOT * 1024; i += 256) {
        int k  = i >> 10;
        int r  = i & 1023;
        int h  = r >> 9;
        int r2 = r & 511;
        int ln = r2 >> 4;
        int sl = r2 & 15;
        int col = h * 16 + (ln & 15);
        int kk  = ((ln >= 16) ? 8 : 0) + (sl & 7) + ((sl >> 3) << 4);
        wl[i] = (_Float16)weight[k * (FDIM * FDIM) + kk * FDIM + col];
    }
    __syncthreads();

    const int lane = threadIdx.x & 31;
    const int wave = threadIdx.x >> 5;
    const int el = lane & 15;
    const int hi = lane >> 4;
    const int stride = gridDim.x * 8;

    const float aw2_lo = attw[32 + el];
    const float aw2_hi = attw[48 + el];

    for (int tile = blockIdx.x * 8 + wave; tile < ETILES; tile += stride) {
        const int tb = tile * 16;
        const int e  = tb + el;                  // lanes L and L+16 share an edge row
        const int src = ecol[e];

        // degree-1 open B-spline basis over 2 dims, K=5
        float p0 = pseudo[(size_t)e * 2 + 0] * (float)(KSZ - 1);
        float p1 = pseudo[(size_t)e * 2 + 1] * (float)(KSZ - 1);
        float f0 = floorf(p0), f1 = floorf(p1);
        float fr0 = p0 - f0, fr1 = p1 - f1;
        int i0 = (int)f0, i1 = (int)f1;
        int c00 = min(max(i0,     0), KSZ - 1);
        int c01 = min(max(i0 + 1, 0), KSZ - 1);
        int c10 = min(max(i1,     0), KSZ - 1);
        int c11 = min(max(i1 + 1, 0), KSZ - 1);
        float bs[4]; int wx[4];
        bs[0] = (1.f - fr0) * (1.f - fr1); wx[0] = c00 + KSZ * c10;
        bs[1] = fr0         * (1.f - fr1); wx[1] = c01 + KSZ * c10;
        bs[2] = (1.f - fr0) * fr1;         wx[2] = c00 + KSZ * c11;
        bs[3] = fr0         * fr1;         wx[3] = c01 + KSZ * c11;

        v16h a = load_a_row(x + (size_t)src * FDIM, hi);

        v8f c0 = {}; v8f c1 = {};
        for (int k = 0; k < KTOT; ++k) {
            float beta = 0.f;
#pragma unroll
            for (int s = 0; s < 4; ++s) beta += (wx[s] == k) ? bs[s] : 0.f;
            if (__any(beta != 0.f)) {            // wave-uniform skip; EXEC stays full
                _Float16 bh = (_Float16)beta;
                v16h as = a * bh;                // fold beta into A rows (pk_mul_f16)
                v16h b0 = *(const v16h*)(wl + k * 1024 + lane * 16);
                v16h b1 = *(const v16h*)(wl + k * 1024 + 512 + lane * 16);
                c0 = __builtin_amdgcn_wmma_f32_16x16x32_f16(false, as, false, b0, (short)0, c0, false, false);
                c1 = __builtin_amdgcn_wmma_f32_16x16x32_f16(false, as, false, b1, (short)0, c1, false, false);
            }
        }

        // D layout: lane el holds column el (c0) / 16+el (c1), rows r+8*hi.
        float myalpha = 0.f;
#pragma unroll
        for (int r = 0; r < 8; ++r) {
            int m = r + hi * 8;
            size_t eo = (size_t)(tb + m) * FDIM;
            msg[eo + el]      = (_Float16)c0[r];      // half-wave-contiguous b16 stores
            msg[eo + 16 + el] = (_Float16)c1[r];
            float t = c0[r] * aw2_lo + c1[r] * aw2_hi;
            t = red16(t);
            if (el == r) myalpha = t;
        }
        if (el < 8) {
            int we = tb + el + hi * 8;
            int trg = erow[we];
            float av = myalpha + rootdot[trg];
            av = (av > 0.f) ? av : NEG_SLOPE * av;   // leaky relu
            alpha[we] = av;
        }
    }
}

// ---------- CSR build: histogram -> exclusive scan -> scatter ----------

__global__ __launch_bounds__(256)
void k_hist(const int* __restrict__ erow, unsigned* __restrict__ deg)
{
    int e = blockIdx.x * 256 + threadIdx.x;
    if (e < NEDGES) atomicAdd(deg + erow[e], 1u);
}

__global__ __launch_bounds__(256)
void k_scan_partial(const unsigned* __restrict__ deg, unsigned* __restrict__ partial)
{
    __shared__ unsigned s[256];
    int b = blockIdx.x;
    int lo = b * CHUNK;
    int n  = min(CHUNK, NNODES - lo);
    unsigned acc = 0;
    for (int i = threadIdx.x; i < n; i += 256) acc += deg[lo + i];
    s[threadIdx.x] = acc;
    __syncthreads();
    for (int off = 128; off > 0; off >>= 1) {
        if (threadIdx.x < off) s[threadIdx.x] += s[threadIdx.x + off];
        __syncthreads();
    }
    if (threadIdx.x == 0) partial[b] = s[0];
}

__global__ __launch_bounds__(256)
void k_scan_base(unsigned* __restrict__ partial)
{
    __shared__ unsigned s[SCAN_BLOCKS];
    s[threadIdx.x] = partial[threadIdx.x];
    __syncthreads();
    if (threadIdx.x == 0) {
        unsigned run = 0;
        for (int i = 0; i < SCAN_BLOCKS; ++i) { unsigned v = s[i]; s[i] = run; run += v; }
    }
    __syncthreads();
    partial[threadIdx.x] = s[threadIdx.x];
}

__global__ __launch_bounds__(256)
void k_scan_write(const unsigned* __restrict__ deg, const unsigned* __restrict__ partial,
                  unsigned* __restrict__ rowptr, unsigned* __restrict__ cursor)
{
    __shared__ unsigned s[CHUNK];            // 391 * 4 B
    int b  = blockIdx.x;
    int lo = b * CHUNK;
    int n  = min(CHUNK, NNODES - lo);
    for (int i = threadIdx.x; i < n; i += 256) s[i] = deg[lo + i];
    __syncthreads();
    if (threadIdx.x == 0) {
        unsigned run = partial[b];
        for (int i = 0; i < n; ++i) { unsigned v = s[i]; s[i] = run; run += v; }
    }
    __syncthreads();
    for (int i = threadIdx.x; i < n; i += 256) {
        rowptr[lo + i] = s[i];
        cursor[lo + i] = s[i];
    }
    if (b == 0 && threadIdx.x == 0) rowptr[NNODES] = (unsigned)NEDGES;
}

__global__ __launch_bounds__(256)
void k_scatter(const int* __restrict__ erow, unsigned* __restrict__ cursor,
               int* __restrict__ esrt)
{
    int e = blockIdx.x * 256 + threadIdx.x;
    if (e < NEDGES) {
        unsigned p = atomicAdd(cursor + erow[e], 1u);
        esrt[p] = e;
    }
}

// ---------- fused per-node softmax + aggregation (no atomics) ----------
// One wave per node; lanes own the 32 output features. msg rows are
// contiguous 64B f16 reads; alpha/esrt loads broadcast across the wave.

__global__ __launch_bounds__(256)
void k_node(const int* __restrict__ esrt, const unsigned* __restrict__ rowptr,
            const float* __restrict__ alpha, const _Float16* __restrict__ msg,
            float* __restrict__ out)
{
    int wid  = (blockIdx.x * 256 + threadIdx.x) >> 5;
    int lane = threadIdx.x & 31;
    if (wid >= NNODES) return;
    int start = (int)rowptr[wid];
    int end   = (int)rowptr[wid + 1];
    if (start == end) return;                 // no edges: out stays x_root + bias

    float m = -3.0e38f;
    for (int i = start; i < end; ++i) m = fmaxf(m, alpha[esrt[i]]);

    float s = 0.f, acc = 0.f;
    for (int i = start; i < end; ++i) {
        int e = esrt[i];
        float ex = __expf(alpha[e] - m);
        s   += ex;
        acc += ex * (float)msg[(size_t)e * FDIM + lane];
    }
    size_t o = (size_t)wid * FDIM + lane;
    out[o] += acc / (s + 1e-16f);             // single owner -> plain RMW
}

// ---------- host launcher ----------

extern "C" void kernel_launch(void* const* d_in, const int* in_sizes, int n_in,
                              void* d_out, int out_size, void* d_ws, size_t ws_size,
                              hipStream_t stream)
{
    const float* x      = (const float*)d_in[0];
    const int*   ei     = (const int*)  d_in[1];   // [2, E]: row then col
    const float* pseudo = (const float*)d_in[2];
    const float* weight = (const float*)d_in[3];
    const float* rootw  = (const float*)d_in[4];
    const float* attw   = (const float*)d_in[5];
    const float* bias   = (const float*)d_in[6];
    float* out = (float*)d_out;

    const int* erow = ei;
    const int* ecol = ei + NEDGES;

    // workspace carve-up (~117 MB)
    unsigned char* w = (unsigned char*)d_ws;
    _Float16* msg    = (_Float16*)w;  w += (size_t)NEDGES * FDIM * sizeof(_Float16);
    float* alpha     = (float*)w;     w += (size_t)NEDGES * sizeof(float);
    float* rootdot   = (float*)w;     w += (size_t)NNODES * sizeof(float);
    unsigned* deg    = (unsigned*)w;  w += (size_t)NNODES * sizeof(unsigned);
    unsigned* rowptr = (unsigned*)w;  w += (size_t)(NNODES + 1) * sizeof(unsigned);
    unsigned* cursor = (unsigned*)w;  w += (size_t)NNODES * sizeof(unsigned);
    unsigned* part   = (unsigned*)w;  w += (size_t)SCAN_BLOCKS * sizeof(unsigned);
    int* esrt        = (int*)w;       w += (size_t)NEDGES * sizeof(int);

    hipMemsetAsync(deg, 0, (size_t)NNODES * sizeof(unsigned), stream);

    // 1) x_root GEMM (WMMA), writes out = x_root + bias and rootdot
    k_xroot<<<(NTILES + 7) / 8, 256, 0, stream>>>(x, rootw, attw, bias, out, rootdot);

    // 2) edge messages (WMMA over 25 kernels) + attention logits
    k_edge_msg<<<2048, 256, 0, stream>>>(x, erow, ecol, pseudo, weight, attw,
                                         rootdot, msg, alpha);

    // 3) CSR of edges by target node
    k_hist<<<(NEDGES + 255) / 256, 256, 0, stream>>>(erow, deg);
    k_scan_partial<<<SCAN_BLOCKS, 256, 0, stream>>>(deg, part);
    k_scan_base<<<1, 256, 0, stream>>>(part);
    k_scan_write<<<SCAN_BLOCKS, 256, 0, stream>>>(deg, part, rowptr, cursor);
    k_scatter<<<(NEDGES + 255) / 256, 256, 0, stream>>>(erow, cursor, esrt);

    // 4) fused segment softmax + weighted aggregation, one wave per node
    k_node<<<(NNODES * 32 + 255) / 256, 256, 0, stream>>>(esrt, rowptr, alpha, msg, out);
}
